// DeepSeekAttention_38886633898647
// MI455X (gfx1250) — matrix operand order repaired and checked by reading, not compiled, not run on hardware
//
#include <hip/hip_runtime.h>

// ---------------------------------------------------------------------------
// DeepSeek MLA attention layer for MI455X (gfx1250), wave32 + WMMA bf16.
// - All GEMM-shaped math on v_wmma_f32_16x16x32_bf16 (fp32 -> bf16 on the fly)
// - Double-buffered LDS, one barrier per K-step, register/async prefetch
// - Pure-bitcopy tile staging uses GLOBAL_LOAD_ASYNC_TO_LDS_B128 (ASYNCcnt)
//   when the toolchain exposes the builtin (guarded, sync-copy fallback).
// ---------------------------------------------------------------------------

typedef __attribute__((ext_vector_type(16))) __bf16 v16bf;
typedef __attribute__((ext_vector_type(8)))  float  v8f;
typedef __attribute__((ext_vector_type(4)))  int    v4i;

#if defined(__has_builtin)
#if __has_builtin(__builtin_amdgcn_global_load_async_to_lds_b128) && \
    __has_builtin(__builtin_amdgcn_s_wait_asynccnt)
#define HAVE_ASYNC 1
#endif
#endif
#ifndef HAVE_ASYNC
#define HAVE_ASYNC 0
#endif

union Frag {
    v16bf v;
    uint4 u[2];
    unsigned short s[16];
};

__device__ __forceinline__ unsigned short bf16c(float f) {
    unsigned int u = __float_as_uint(f);
    u += 0x7FFFu + ((u >> 16) & 1u);              // round-to-nearest-even
    return (unsigned short)(u >> 16);
}
__device__ __forceinline__ unsigned int bf16pack(float a, float b) {
    return (unsigned int)bf16c(a) | ((unsigned int)bf16c(b) << 16);
}
__device__ __forceinline__ float bf2f(unsigned short h) {
    return __uint_as_float(((unsigned int)h) << 16);
}
__device__ __forceinline__ v8f wmma_bf16(const Frag& a, const Frag& b, v8f c) {
    return __builtin_amdgcn_wmma_f32_16x16x32_bf16(false, a.v, false, b.v,
                                                   (short)0, c, false, false);
}

// 16-byte global -> LDS copy; async (ASYNCcnt) when available.
__device__ __forceinline__ void async_cp16(const void* g, void* l) {
#if HAVE_ASYNC
    __builtin_amdgcn_global_load_async_to_lds_b128(
        (__attribute__((address_space(1))) v4i*)g,
        (__attribute__((address_space(3))) v4i*)l, 0, 0);
#else
    *(uint4*)l = *(const uint4*)g;
#endif
}
__device__ __forceinline__ void async_wait() {
#if HAVE_ASYNC
    __builtin_amdgcn_s_wait_asynccnt(0);
#endif
}

// ---------------------------------------------------------------------------
// Tiled GEMM: C(MxN) = A(MxK,row-major) * B(KxN,row-major fp32).
// BM=128 BN=128 BK=32, 256 threads = 8 waves (2x4), wave tile 64x32.
// Double-buffered LDS, next tile prefetched (regs / async) during compute.
// EPI 0: f32 row-major out0.  EPI 1: bf16 head-major (192/head).
// EPI 2: split 256/head -> Kp (stride 192) + Vp (stride 128).
// ABF16: A operand is bf16 row-major (staged via async LDS copy).
// ---------------------------------------------------------------------------
template<int EPI, bool ABF16>
__global__ __launch_bounds__(256) void gemm_wmma_k(
    const void* __restrict__ Aptr, const float* __restrict__ B,
    void* __restrict__ out0, void* __restrict__ out1,
    int M, int N, int K, int S)
{
    __shared__ unsigned int apack[2][128][20];   // [buf][m][kpair]
    __shared__ unsigned int bpack[2][128][20];   // [buf][n][kpair]

    const int tid  = threadIdx.x;
    const int m0   = blockIdx.y * 128;
    const int n0   = blockIdx.x * 128;
    const int wave = tid >> 5, lane = tid & 31;
    const int wm   = wave >> 2, wn = wave & 3;
    const int lh   = lane >> 4, ll = lane & 15;

    v8f acc[4][2];
    #pragma unroll
    for (int mt = 0; mt < 4; ++mt)
        #pragma unroll
        for (int nt = 0; nt < 2; ++nt)
            #pragma unroll
            for (int r = 0; r < 8; ++r) acc[mt][nt][r] = 0.0f;

    const int arow = tid >> 1, ahalf = tid & 1;         // A: 2 thr/row, 16 elems
    const int bp   = tid >> 4, bnb = (tid & 15) * 8;    // B: k-pair bp, 8 cols

    float4 fa[4];                 // fp32 A prefetch regs
    float  rb0[8], rb1[8];        // B prefetch regs

    auto loadA = [&](int k0, int buf) {
        if constexpr (ABF16) {
            const unsigned short* A = (const unsigned short*)Aptr;
            const unsigned short* g = A + (size_t)(m0 + arow) * K + k0 + ahalf * 16;
            async_cp16(g,     &apack[buf][arow][ahalf * 8]);
            async_cp16(g + 8, &apack[buf][arow][ahalf * 8 + 4]);
        } else {
            const float* A = (const float*)Aptr;
            const float4* src =
                (const float4*)(A + (size_t)(m0 + arow) * K + k0 + ahalf * 16);
            fa[0] = src[0]; fa[1] = src[1]; fa[2] = src[2]; fa[3] = src[3];
        }
    };
    auto storeA = [&](int buf) {
        if constexpr (!ABF16) {
            unsigned int* dst = &apack[buf][arow][ahalf * 8];
            #pragma unroll
            for (int i = 0; i < 4; ++i) {
                dst[2 * i]     = bf16pack(fa[i].x, fa[i].y);
                dst[2 * i + 1] = bf16pack(fa[i].z, fa[i].w);
            }
        }
    };
    auto loadB = [&](int k0) {
        const float* B0 = B + (size_t)(k0 + 2 * bp) * N + n0 + bnb;
        const float* B1 = B0 + N;
        if (n0 + bnb + 7 < N) {
            float4 a0 = *(const float4*)B0, a1 = *(const float4*)(B0 + 4);
            float4 c0 = *(const float4*)B1, c1 = *(const float4*)(B1 + 4);
            rb0[0]=a0.x; rb0[1]=a0.y; rb0[2]=a0.z; rb0[3]=a0.w;
            rb0[4]=a1.x; rb0[5]=a1.y; rb0[6]=a1.z; rb0[7]=a1.w;
            rb1[0]=c0.x; rb1[1]=c0.y; rb1[2]=c0.z; rb1[3]=c0.w;
            rb1[4]=c1.x; rb1[5]=c1.y; rb1[6]=c1.z; rb1[7]=c1.w;
        } else {
            #pragma unroll
            for (int i = 0; i < 8; ++i) {
                bool ok = (n0 + bnb + i) < N;
                rb0[i] = ok ? B0[i] : 0.0f;
                rb1[i] = ok ? B1[i] : 0.0f;
            }
        }
    };
    auto storeB = [&](int buf) {
        #pragma unroll
        for (int i = 0; i < 8; ++i)
            bpack[buf][bnb + i][bp] = bf16pack(rb0[i], rb1[i]);
    };

    int buf = 0;
    loadA(0, 0);
    loadB(0);

    for (int k0 = 0; k0 < K; k0 += 32) {
        storeA(buf);
        storeB(buf);
        async_wait();
        __syncthreads();

        const int kn = k0 + 32;
        if (kn < K) {            // prefetch next tile during compute
            loadA(kn, buf ^ 1);
            loadB(kn);
        }

        Frag af[4], bfr[2];
        #pragma unroll
        for (int mt = 0; mt < 4; ++mt) {
            const unsigned int* p = &apack[buf][wm * 64 + mt * 16 + ll][lh * 4];
            af[mt].u[0] = *(const uint4*)p;
            af[mt].u[1] = *(const uint4*)(p + 8);
        }
        #pragma unroll
        for (int nt = 0; nt < 2; ++nt) {
            const unsigned int* p = &bpack[buf][wn * 32 + nt * 16 + ll][lh * 4];
            bfr[nt].u[0] = *(const uint4*)p;
            bfr[nt].u[1] = *(const uint4*)(p + 8);
        }
        #pragma unroll
        for (int mt = 0; mt < 4; ++mt)
            #pragma unroll
            for (int nt = 0; nt < 2; ++nt)
                acc[mt][nt] = wmma_bf16(af[mt], bfr[nt], acc[mt][nt]);
        buf ^= 1;
    }

    #pragma unroll
    for (int mt = 0; mt < 4; ++mt)
        #pragma unroll
        for (int nt = 0; nt < 2; ++nt)
            #pragma unroll
            for (int r = 0; r < 8; ++r) {
                int m = m0 + wm * 64 + mt * 16 + r + 8 * lh;
                int n = n0 + wn * 32 + nt * 16 + ll;
                if (n >= N) continue;
                float val = acc[mt][nt][r];
                if constexpr (EPI == 0) {
                    ((float*)out0)[(size_t)m * N + n] = val;
                } else if constexpr (EPI == 1) {
                    int h = n / 192, d = n - h * 192;
                    ((unsigned short*)out0)[((size_t)h * S + m) * 192 + d] = bf16c(val);
                } else {
                    int h = n / 256, d = n - h * 256;
                    if (d < 128)
                        ((unsigned short*)out0)[((size_t)h * S + m) * 192 + d] = bf16c(val);
                    else
                        ((unsigned short*)out1)[((size_t)h * S + m) * 128 + (d - 128)] = bf16c(val);
                }
            }
}

// ---------------------------------------------------------------------------
// Row LayerNorm: out = (x - mean)/sqrt(var + 1e-5) * g, one block per row.
// ---------------------------------------------------------------------------
__global__ __launch_bounds__(256) void ln_k(const float* __restrict__ in,
                                            const float* __restrict__ g,
                                            float* __restrict__ out,
                                            int W, int ldin, int ldout)
{
    __shared__ float red[256];
    const int row = blockIdx.x, t = threadIdx.x;
    const float* x = in + (size_t)row * ldin;
    float s = 0.0f;
    for (int j = t; j < W; j += 256) s += x[j];
    red[t] = s; __syncthreads();
    for (int d = 128; d > 0; d >>= 1) { if (t < d) red[t] += red[t + d]; __syncthreads(); }
    float mean = red[0] / (float)W;
    __syncthreads();
    float q = 0.0f;
    for (int j = t; j < W; j += 256) { float dd = x[j] - mean; q += dd * dd; }
    red[t] = q; __syncthreads();
    for (int d = 128; d > 0; d >>= 1) { if (t < d) red[t] += red[t + d]; __syncthreads(); }
    float rstd = rsqrtf(red[0] / (float)W + 1e-5f);
    float* o = out + (size_t)row * ldout;
    for (int j = t; j < W; j += 256) o[j] = (x[j] - mean) * rstd * g[j];
}

// RoPE for shared k_pe: reads ckv_raw[:,512:576] fp32, writes roped fp32.
__global__ void rope_kpe_k(const float* __restrict__ ckv_raw,
                           const int* __restrict__ pos_ids,
                           float* __restrict__ kpe)
{
    const int row = blockIdx.x, j = threadIdx.x;      // 64 threads
    const float* base = ckv_raw + (size_t)row * 576 + 512;
    float x  = base[j];
    float xp = base[(j + 32) & 63];
    float ang = (float)pos_ids[row] * __powf(10000.0f, -((float)(j & 31)) / 32.0f);
    float c = __cosf(ang), s = __sinf(ang);
    kpe[(size_t)row * 64 + j] = (j < 32) ? (x * c - xp * s) : (x * c + xp * s);
}

// RoPE for q_pe in-place on bf16 head-major Q' (dims 128..191 of each head).
__global__ void rope_q_k(unsigned short* __restrict__ Qp,
                         const int* __restrict__ pos_ids)
{
    const int row = blockIdx.x, h = blockIdx.y, j = threadIdx.x;   // 64 threads
    unsigned short* base = Qp + ((size_t)h * 1024 + row) * 192 + 128;
    float x  = bf2f(base[j]);
    float xp = bf2f(base[(j + 32) & 63]);
    __syncthreads();                                 // reads before in-place writes
    float ang = (float)pos_ids[row] * __powf(10000.0f, -((float)(j & 31)) / 32.0f);
    float c = __cosf(ang), s = __sinf(ang);
    base[j] = bf16c((j < 32) ? (x * c - xp * s) : (x * c + xp * s));
}

// Broadcast roped k_pe into every head's K' pe slot (bf16).
__global__ __launch_bounds__(256) void kpe_bcast_k(const float* __restrict__ kpe,
                                                   unsigned short* __restrict__ Kp)
{
    size_t idx = (size_t)blockIdx.x * 256 + threadIdx.x;   // 128*1024*64 total
    int h   = (int)(idx >> 16);
    int rem = (int)(idx & 0xFFFF);
    int row = rem >> 6, j = rem & 63;
    Kp[((size_t)h * 1024 + row) * 192 + 128 + j] = bf16c(kpe[(size_t)row * 64 + j]);
}

// ---------------------------------------------------------------------------
// Flash attention: block = (128 queries x 1 head), 8 waves x 16 rows.
// Q'(192) persistent in VGPR fragments; K' tile staged via async LDS copies,
// V tile pair-packed; double-buffered, one barrier per 32-key step; online
// softmax with 16-lane shuffle reductions; P routed through per-wave LDS.
// ---------------------------------------------------------------------------
__global__ __launch_bounds__(256) void attn_k(
    const unsigned short* __restrict__ Qp,
    const unsigned short* __restrict__ Kp,
    const unsigned short* __restrict__ Vp,
    const float* __restrict__ mask,
    unsigned short* __restrict__ Obf,
    float scale)
{
    __shared__ unsigned int kT[2][32][100];   // [buf][key][kpair 0..95]
    __shared__ unsigned int vT[2][128][20];   // [buf][dim][kpair 0..15]
    __shared__ float pS[8][16][32];           // per-wave P scratch (C->A layout)

    const int h  = blockIdx.y;
    const int q0 = blockIdx.x * 128;
    const int tid = threadIdx.x;
    const int wave = tid >> 5, lane = tid & 31;
    const int lh = lane >> 4, ll = lane & 15;

    // persistent Q fragments (row m = ll of this wave's 16-row tile)
    Frag qf[6];
    {
        const unsigned int* qrow =
            (const unsigned int*)(Qp + ((size_t)h * 1024 + q0 + wave * 16 + ll) * 192);
        #pragma unroll
        for (int s = 0; s < 6; ++s) {
            qf[s].u[0] = *(const uint4*)(qrow + 16 * s + lh * 4);
            qf[s].u[1] = *(const uint4*)(qrow + 16 * s + lh * 4 + 8);
        }
    }

    v8f o[8];
    #pragma unroll
    for (int ot = 0; ot < 8; ++ot)
        #pragma unroll
        for (int r = 0; r < 8; ++r) o[ot][r] = 0.0f;
    float mrun[8], lrun[8];
    #pragma unroll
    for (int r = 0; r < 8; ++r) { mrun[r] = -1e30f; lrun[r] = 0.0f; }

    const int krow = tid >> 3, kcb = (tid & 7) * 12;  // K' tile: 8 thr/row, 12 uints
    const int vp   = tid >> 4, vnb = (tid & 15) * 8;  // V tile: pair vp, 8 dims

    auto stage = [&](int kb, int buf) {
        // K' rows are pair-contiguous -> pure bitcopy, async to LDS
        const unsigned int* kg =
            (const unsigned int*)(Kp + ((size_t)h * 1024 + kb + krow) * 192) + kcb;
        async_cp16(kg,     &kT[buf][krow][kcb]);
        async_cp16(kg + 4, &kT[buf][krow][kcb + 4]);
        async_cp16(kg + 8, &kT[buf][krow][kcb + 8]);
        // V needs key-row pair interleave -> manual pack
        const unsigned short* v0 = Vp + ((size_t)h * 1024 + kb + 2 * vp) * 128 + vnb;
        uint4 a = *(const uint4*)v0;
        uint4 b = *(const uint4*)(v0 + 128);
        const unsigned short* as = (const unsigned short*)&a;
        const unsigned short* bs = (const unsigned short*)&b;
        #pragma unroll
        for (int i = 0; i < 8; ++i)
            vT[buf][vnb + i][vp] = (unsigned int)as[i] | ((unsigned int)bs[i] << 16);
    };

    int buf = 0;
    stage(0, 0);

    for (int kb = 0; kb < 1024; kb += 32) {
        async_wait();
        __syncthreads();
        if (kb + 32 < 1024) stage(kb + 32, buf ^ 1);   // overlap with compute

        // ---- S = Q' K'^T (two 16-key tiles, 6 k-steps) ----
        v8f sA, sB;
        #pragma unroll
        for (int r = 0; r < 8; ++r) { sA[r] = 0.0f; sB[r] = 0.0f; }
        #pragma unroll
        for (int s = 0; s < 6; ++s) {
            Frag bk0, bk1;
            const unsigned int* p0 = &kT[buf][ll][16 * s + lh * 4];
            const unsigned int* p1 = &kT[buf][16 + ll][16 * s + lh * 4];
            bk0.u[0] = *(const uint4*)p0;  bk0.u[1] = *(const uint4*)(p0 + 8);
            bk1.u[0] = *(const uint4*)p1;  bk1.u[1] = *(const uint4*)(p1 + 8);
            sA = wmma_bf16(qf[s], bk0, sA);
            sB = wmma_bf16(qf[s], bk1, sB);
        }

        // ---- online softmax (rows r+8*lh, per-half 16-lane reductions) ----
        float pa[8], pb[8], corr[8];
        #pragma unroll
        for (int r = 0; r < 8; ++r) {
            size_t qrow = (size_t)(q0 + wave * 16 + r + 8 * lh) * 1024 + kb;
            float va = sA[r] * scale + mask[qrow + ll];
            float vb = sB[r] * scale + mask[qrow + 16 + ll];
            float t = fmaxf(va, vb);
            #pragma unroll
            for (int d = 1; d < 16; d <<= 1) t = fmaxf(t, __shfl_xor(t, d, 32));
            float mn = fmaxf(mrun[r], t);
            corr[r] = __expf(mrun[r] - mn);
            mrun[r] = mn;
            pa[r] = __expf(va - mn);
            pb[r] = __expf(vb - mn);
            float su = pa[r] + pb[r];
            #pragma unroll
            for (int d = 1; d < 16; d <<= 1) su += __shfl_xor(su, d, 32);
            lrun[r] = lrun[r] * corr[r] + su;
        }
        #pragma unroll
        for (int ot = 0; ot < 8; ++ot)
            #pragma unroll
            for (int r = 0; r < 8; ++r) o[ot][r] *= corr[r];

        // ---- P: C-layout -> per-wave LDS -> A-layout fragment ----
        #pragma unroll
        for (int r = 0; r < 8; ++r) {
            pS[wave][r + 8 * lh][ll]      = pa[r];
            pS[wave][r + 8 * lh][16 + ll] = pb[r];
        }
        Frag pf;                       // same-wave DS ordering covers pS
        #pragma unroll
        for (int v = 0; v < 8; ++v) {
            int kk = (v & 3) * 2 + lh * 8 + (v >> 2) * 16;
            float2 f = *(const float2*)&pS[wave][ll][kk];
            pf.s[2 * v]     = bf16c(f.x);
            pf.s[2 * v + 1] = bf16c(f.y);
        }

        // ---- O += P V ----
        #pragma unroll
        for (int ot = 0; ot < 8; ++ot) {
            Frag bv;
            const unsigned int* p = &vT[buf][ot * 16 + ll][lh * 4];
            bv.u[0] = *(const uint4*)p;
            bv.u[1] = *(const uint4*)(p + 8);
            o[ot] = wmma_bf16(pf, bv, o[ot]);
        }
        buf ^= 1;
    }

    // ---- finalize: O / l, write bf16 [seq][H*DV] ----
    float inv[8];
    #pragma unroll
    for (int r = 0; r < 8; ++r) inv[r] = 1.0f / lrun[r];
    #pragma unroll
    for (int ot = 0; ot < 8; ++ot)
        #pragma unroll
        for (int r = 0; r < 8; ++r) {
            int row = q0 + wave * 16 + r + 8 * lh;
            int col = h * 128 + ot * 16 + ll;
            Obf[(size_t)row * 16384 + col] = bf16c(o[ot][r] * inv[r]);
        }
}

// ---------------------------------------------------------------------------
extern "C" void kernel_launch(void* const* d_in, const int* in_sizes, int n_in,
                              void* d_out, int out_size, void* d_ws, size_t ws_size,
                              hipStream_t stream)
{
    (void)in_sizes; (void)n_in; (void)out_size; (void)ws_size;
    const float* hs     = (const float*)d_in[0];   // [1024][5120]
    const float* mask   = (const float*)d_in[1];   // [1024][1024]
    const int*   pos    = (const int*)  d_in[2];   // [1024]
    const float* W_qa   = (const float*)d_in[3];   // [5120][1536]
    const float* g_qa   = (const float*)d_in[4];   // [1536]
    const float* W_qb   = (const float*)d_in[5];   // [1536][24576]
    const float* W_kva  = (const float*)d_in[6];   // [5120][576]
    const float* g_kva  = (const float*)d_in[7];   // [512]
    const float* W_kvb  = (const float*)d_in[8];   // [512][32768]
    const float* W_o    = (const float*)d_in[9];   // [16384][5120]
    float* out = (float*)d_out;                    // [1024][5120]

    // workspace layout (~185 MB)
    char* ws = (char*)d_ws;
    float*          qa_raw  = (float*)ws;          ws += (size_t)1024 * 1536 * 4;
    float*          qa_ln   = (float*)ws;          ws += (size_t)1024 * 1536 * 4;
    float*          ckv_raw = (float*)ws;          ws += (size_t)1024 * 576 * 4;
    float*          ckv_ln  = (float*)ws;          ws += (size_t)1024 * 512 * 4;
    float*          kpe     = (float*)ws;          ws += (size_t)1024 * 64 * 4;
    unsigned short* Qp      = (unsigned short*)ws; ws += (size_t)128 * 1024 * 192 * 2;
    unsigned short* Kp      = (unsigned short*)ws; ws += (size_t)128 * 1024 * 192 * 2;
    unsigned short* Vp      = (unsigned short*)ws; ws += (size_t)128 * 1024 * 128 * 2;
    unsigned short* Attn    = (unsigned short*)ws; ws += (size_t)1024 * 16384 * 2;

    // 1) qa_raw = hs @ W_qa                         (1024x1536, K=5120)
    gemm_wmma_k<0, false><<<dim3(12, 8), 256, 0, stream>>>(
        hs, W_qa, qa_raw, nullptr, 1024, 1536, 5120, 1024);
    // 2) qa_ln = LN(qa_raw, g_qa)
    ln_k<<<1024, 256, 0, stream>>>(qa_raw, g_qa, qa_ln, 1536, 1536, 1536);
    // 3) ckv_raw = hs @ W_kva                       (1024x576, K=5120)
    gemm_wmma_k<0, false><<<dim3(5, 8), 256, 0, stream>>>(
        hs, W_kva, ckv_raw, nullptr, 1024, 576, 5120, 1024);
    // 4) ckv_ln = LN(ckv_raw[:, :512], g_kva)
    ln_k<<<1024, 256, 0, stream>>>(ckv_raw, g_kva, ckv_ln, 512, 576, 512);
    // 5) kpe = rope(ckv_raw[:, 512:576])
    rope_kpe_k<<<1024, 64, 0, stream>>>(ckv_raw, pos, kpe);
    // 6) Q' (bf16, head-major 192/d) = qa_ln @ W_qb (1024x24576, K=1536)
    gemm_wmma_k<1, false><<<dim3(192, 8), 256, 0, stream>>>(
        qa_ln, W_qb, Qp, nullptr, 1024, 24576, 1536, 1024);
    // 7) rope q_pe in place
    rope_q_k<<<dim3(1024, 128), 64, 0, stream>>>(Qp, pos);
    // 8) K'(nope) + V (bf16, head-major) = ckv_ln @ W_kvb  (1024x32768, K=512)
    gemm_wmma_k<2, false><<<dim3(256, 8), 256, 0, stream>>>(
        ckv_ln, W_kvb, Kp, Vp, 1024, 32768, 512, 1024);
    // 9) broadcast roped k_pe into every head's K'
    kpe_bcast_k<<<32768, 256, 0, stream>>>(kpe, Kp);
    // 10) flash attention -> Attn bf16 [1024][16384]
    const float scale = 0.07216878364870323f;      // 1/sqrt(192)
    attn_k<<<dim3(8, 128), 256, 0, stream>>>(Qp, Kp, Vp, mask, Attn, scale);
    // 11) out = Attn @ W_o                          (1024x5120, K=16384)
    gemm_wmma_k<0, true><<<dim3(40, 8), 256, 0, stream>>>(
        Attn, W_o, out, nullptr, 1024, 5120, 16384, 1024);
}
